// GAT2_6605659701637
// MI455X (gfx1250) — compile-verified, matrix operand order checked by reading, not run on hardware
//
#include <hip/hip_runtime.h>
#include <hip/hip_bf16.h>

// ---------------------------------------------------------------------------
// GAT pipeline for MI455X (gfx1250, wave32, WMMA).
// GEMMs run on v_wmma_f32_16x16x32_bf16 (fp32 accumulate) with double-buffered
// LDS tiles; the A-operand stream uses GLOBAL_LOAD_ASYNC_TO_LDS_B128 (ASYNCcnt)
// when the toolchain exposes the builtin. Edge phase uses fp32 atomics.
// ---------------------------------------------------------------------------

static constexpr int NN   = 50000;   // nodes
static constexpr int EE   = 800000;  // directed edges (before self loops)
static constexpr int ETOT = EE + NN; // edges incl. self loops
static constexpr int FIN  = 2613;    // raw feature dim
static constexpr int DD   = 256;     // hidden dim
static constexpr int CH   = 250;     // stats reduction chunks (250*200 == 50000)
static constexpr int RPC  = 200;     // rows per chunk

typedef __bf16 bf16_t;
typedef bf16_t   v16bf __attribute__((ext_vector_type(16)));
typedef float    v8f   __attribute__((ext_vector_type(8)));
typedef unsigned v4u   __attribute__((ext_vector_type(4)));

union Frag { v16bf v; v4u q[2]; };
static_assert(sizeof(Frag) == 32, "frag size");

__device__ __forceinline__ unsigned short f2bf(float f) {
  unsigned u = __float_as_uint(f);
  u += 0x7FFFu + ((u >> 16) & 1u);       // round-to-nearest-even
  return (unsigned short)(u >> 16);
}
__device__ __forceinline__ unsigned f2ord(float f) {  // order-preserving map
  unsigned u = __float_as_uint(f);
  return (u & 0x80000000u) ? ~u : (u | 0x80000000u);
}
__device__ __forceinline__ float ord2f(unsigned u) {
  return __uint_as_float((u & 0x80000000u) ? (u & 0x7FFFFFFFu) : ~u);
}

// --- CDNA5 async global->LDS copy (ASYNCcnt path), probe-guarded ------------
#if defined(__has_builtin)
#if __has_builtin(__builtin_amdgcn_global_load_async_to_lds_b128)
#define HAVE_ASYNC_COPY 1
#endif
#endif
#ifndef HAVE_ASYNC_COPY
#define HAVE_ASYNC_COPY 0
#endif

#if HAVE_ASYNC_COPY
// Per hipcc diagnostic, param 1 is: int __attribute__((vector_size(16))) __device__*
// (AS1 = global). Param 2 mirrors it in AS3 (LDS), then imm offset + imm cpol.
typedef int v4i_gcc __attribute__((vector_size(16)));
typedef __attribute__((address_space(1))) v4i_gcc* g_v4i_ptr;
typedef __attribute__((address_space(3))) v4i_gcc* l_v4i_ptr;
__device__ __forceinline__ void async_cp16(void* lds, const void* g) {
  __builtin_amdgcn_global_load_async_to_lds_b128((g_v4i_ptr)g, (l_v4i_ptr)lds,
                                                 0, 0);
}
__device__ __forceinline__ void wait_async_le2() {
#if __has_builtin(__builtin_amdgcn_s_wait_asynccnt)
  __builtin_amdgcn_s_wait_asynccnt(2);
#else
  asm volatile("s_wait_asynccnt 0x2" ::: "memory");
#endif
}
__device__ __forceinline__ void wait_async_le0() {
#if __has_builtin(__builtin_amdgcn_s_wait_asynccnt)
  __builtin_amdgcn_s_wait_asynccnt(0);
#else
  asm volatile("s_wait_asynccnt 0x0" ::: "memory");
#endif
}
#endif

// ---------------------------------------------------------------------------
// fp32 -> bf16 bit conversion (weight prep)
// ---------------------------------------------------------------------------
__global__ __launch_bounds__(256) void f32_to_bf16_k(const float* __restrict__ in,
                                                     unsigned short* __restrict__ out,
                                                     int n) {
  int i = blockIdx.x * 256 + threadIdx.x;
  if (i < n) out[i] = f2bf(in[i]);
}

// ---------------------------------------------------------------------------
// Tiled WMMA GEMM: C[M,256] = A[M,K] * Wb[K,256] (+ bias)
// A is fp32 (ABF=false) or bf16 bits (ABF=true); Wb is bf16 bits.
// Block = 256 threads (8 waves). Tile: BM=64, BN=256 (full), BK=64.
// Double-buffered LDS; bf16-A tiles stream via async copies when available.
// Wave w owns column tiles {2w, 2w+1}; 4 row subtiles -> 8 f32 accumulators.
// ---------------------------------------------------------------------------
template <bool ABF>
__global__ __launch_bounds__(256) void gemm_kernel(const void* __restrict__ Aptr,
                                                   const unsigned short* __restrict__ Wb,
                                                   const float* __restrict__ bias,
                                                   float* __restrict__ C,
                                                   int M, int K) {
  constexpr int BM = 64, BK = 64, LDK = BK + 8;  // pad keeps 16B align, skews banks
  __shared__ unsigned short As[2][BM][LDK];   // [buf][m][k]
  __shared__ unsigned short Bs[2][256][LDK];  // [buf][n][k] (transposed)

  const int tid  = threadIdx.x;
  const int wave = tid >> 5;
  const int lane = tid & 31;
  const int row0 = blockIdx.x * BM;
  const int half = lane >> 4;   // 0: lanes 0-15, 1: lanes 16-31
  const int lidx = lane & 15;
  const int kb   = half * 8;    // K sub-offset per documented 16-bit layout

  const v8f vzero = {0.f, 0.f, 0.f, 0.f, 0.f, 0.f, 0.f, 0.f};
  v8f acc[4][2];
  for (int i = 0; i < 4; ++i)
    for (int j = 0; j < 2; ++j) acc[i][j] = vzero;

  const float*          Af = (const float*)Aptr;
  const unsigned short* Ab = (const unsigned short*)Aptr;

  auto stageB = [&](int buf, int kk) {
    for (int e = tid; e < BK * 256; e += 256) {
      int k = e >> 8, n = e & 255;
      int gk = kk + k;
      Bs[buf][n][k] = (gk < K) ? Wb[(size_t)gk * 256 + n] : (unsigned short)0;
    }
  };
  auto stageA = [&](int buf, int kk) {
#if HAVE_ASYNC_COPY
    if (ABF) {
      // 512 x 16B chunks, 2 per thread, ASYNCcnt-tracked DMA into LDS.
      // bf16 path always has K == DD (exact multiple of BK); rows past M read
      // harmless in-workspace bytes and their outputs are never stored.
      for (int j = 0; j < 2; ++j) {
        int c  = tid + 256 * j;
        int m  = c >> 3;              // 8 chunks (128B) per 64-elem row
        int cj = (c & 7) * 8;         // bf16 element offset within row
        const unsigned short* g = Ab + (size_t)(row0 + m) * K + (kk + cj);
        async_cp16(&As[buf][m][cj], g);
      }
      return;
    }
#endif
    for (int e = tid; e < BM * BK; e += 256) {
      int m = e >> 6, k = e & 63;
      int gr = row0 + m, gk = kk + k;
      unsigned short v = 0;
      if (gr < M && gk < K)
        v = ABF ? Ab[(size_t)gr * K + gk] : f2bf(Af[(size_t)gr * K + gk]);
      As[buf][m][k] = v;
    }
  };

  const int nIter = (K + BK - 1) / BK;
  stageA(0, 0);
  stageB(0, 0);

  for (int it = 0; it < nIter; ++it) {
    const int  kk      = it * BK;
    const int  cur     = it & 1;
    const int  nxt     = cur ^ 1;
    const bool hasNext = (it + 1) < nIter;

    if (hasNext) {                       // prefetch next tiles into other buffer
      stageA(nxt, kk + BK);
      stageB(nxt, kk + BK);
    }
#if HAVE_ASYNC_COPY
    if (ABF) {                           // own async A-copies for 'cur' done
      if (hasNext) wait_async_le2();
      else         wait_async_le0();
    }
#endif
    __syncthreads();

    for (int ks = 0; ks < BK; ks += 32) {
      Frag bq[2];
      for (int t = 0; t < 2; ++t) {
        int n = wave * 32 + t * 16 + lidx;
        bq[t].q[0] = *(const v4u*)&Bs[cur][n][ks + kb];
        bq[t].q[1] = *(const v4u*)&Bs[cur][n][ks + 16 + kb];
      }
      for (int rt = 0; rt < 4; ++rt) {
        Frag aq;
        int m = rt * 16 + lidx;
        aq.q[0] = *(const v4u*)&As[cur][m][ks + kb];
        aq.q[1] = *(const v4u*)&As[cur][m][ks + 16 + kb];
        for (int t = 0; t < 2; ++t)
          acc[rt][t] = __builtin_amdgcn_wmma_f32_16x16x32_bf16(
              false, aq.v, false, bq[t].v, (short)0, acc[rt][t], false, false);
      }
    }
    __syncthreads();                     // compute done before buffer reuse
  }

  // --- epilogue: D[m][n] per documented C/D layout; optional bias ---
  for (int rt = 0; rt < 4; ++rt) {
    for (int t = 0; t < 2; ++t) {
      int col = wave * 32 + t * 16 + lidx;
      float bv = bias ? bias[col] : 0.f;
      for (int r = 0; r < 8; ++r) {
        int row = row0 + rt * 16 + r + 8 * half;
        if (row < M) C[(size_t)row * 256 + col] = acc[rt][t][r] + bv;
      }
    }
  }
}

// ---------------------------------------------------------------------------
// BatchNorm stats: deterministic two-stage column reduction over N rows
// ---------------------------------------------------------------------------
__global__ __launch_bounds__(256) void col_stats_partial(const float* __restrict__ pre,
                                                         float* __restrict__ psum,
                                                         float* __restrict__ psq) {
  int c = blockIdx.x, d = threadIdx.x;
  int r0 = c * RPC, r1 = r0 + RPC;
  float s = 0.f, q = 0.f;
  for (int r = r0; r < r1; ++r) {
    float v = pre[(size_t)r * 256 + d];
    s += v; q += v * v;
  }
  psum[c * 256 + d] = s;
  psq[c * 256 + d]  = q;
}

__global__ __launch_bounds__(256) void col_stats_final(const float* __restrict__ psum,
                                                       const float* __restrict__ psq,
                                                       const float* __restrict__ gamma,
                                                       const float* __restrict__ beta,
                                                       float* __restrict__ scale,
                                                       float* __restrict__ shift) {
  int d = threadIdx.x;
  float s = 0.f, q = 0.f;
  for (int c = 0; c < CH; ++c) { s += psum[c * 256 + d]; q += psq[c * 256 + d]; }
  float mean = s * (1.f / NN);
  float var  = q * (1.f / NN) - mean * mean;
  float sc   = gamma[d] * rsqrtf(var + 1e-5f);
  scale[d] = sc;
  shift[d] = beta[d] - mean * sc;
}

// BN-apply + ELU; writes fp32 (optional) and bf16 (optional) copies
__global__ __launch_bounds__(256) void bn_elu(const float* __restrict__ pre,
                                              const float* __restrict__ scale,
                                              const float* __restrict__ shift,
                                              float* __restrict__ outf,
                                              unsigned short* __restrict__ outb,
                                              int total) {
  int i = blockIdx.x * 256 + threadIdx.x;
  if (i >= total) return;
  int d = i & 255;
  float y = pre[i] * scale[d] + shift[d];
  y = (y > 0.f) ? y : expm1f(y);
  if (outf) outf[i] = y;
  if (outb) outb[i] = f2bf(y);
}

// ---------------------------------------------------------------------------
// GAT: per-node attention dots (one wave per node) + per-node init
// ---------------------------------------------------------------------------
__global__ __launch_bounds__(256) void dots_kernel(const float* __restrict__ xl,
                                                   const float* __restrict__ att_s,
                                                   const float* __restrict__ att_d,
                                                   float* __restrict__ a_s,
                                                   float* __restrict__ a_d,
                                                   unsigned* __restrict__ m_bits,
                                                   float* __restrict__ denom) {
  int wave = threadIdx.x >> 5, lane = threadIdx.x & 31;
  int n = blockIdx.x * 8 + wave;
  if (n >= NN) return;
  const float* row = xl + (size_t)n * 256;
  float ss = 0.f, sd = 0.f;
  for (int j = 0; j < 8; ++j) {
    int d = lane + 32 * j;
    float v = row[d];
    ss += v * att_s[d];
    sd += v * att_d[d];
  }
  for (int off = 16; off; off >>= 1) {
    ss += __shfl_xor(ss, off);
    sd += __shfl_xor(sd, off);
  }
  if (lane == 0) {
    a_s[n] = ss; a_d[n] = sd;
    m_bits[n] = 0u;     // ordered encoding below any real score
    denom[n]  = 0.f;
  }
}

// Pre-fill aggregation buffer with broadcast bias (folds "+ bias")
__global__ __launch_bounds__(256) void fill_bias(float* __restrict__ out,
                                                 const float* __restrict__ bias,
                                                 int total) {
  int i = blockIdx.x * 256 + threadIdx.x;
  if (i < total) out[i] = bias[i & 255];
}

// Pass 1: leaky-relu score + segment max via ordered-uint atomicMax
__global__ __launch_bounds__(256) void edge_max(const int* __restrict__ esrc,
                                                const int* __restrict__ edst,
                                                const float* __restrict__ a_s,
                                                const float* __restrict__ a_d,
                                                float* __restrict__ e_out,
                                                unsigned* __restrict__ m_bits) {
  int i = blockIdx.x * 256 + threadIdx.x;
  if (i >= ETOT) return;
  int s, t;
  if (i < EE) { s = esrc[i]; t = edst[i]; } else { s = t = i - EE; }
  float e = a_s[s] + a_d[t];
  e = (e > 0.f) ? e : 0.2f * e;           // leaky_relu, slope 0.2
  e_out[i] = e;
  atomicMax(&m_bits[t], f2ord(e));
}

// Pass 2: w = exp(e - max); segment sum of w
__global__ __launch_bounds__(256) void edge_expsum(const int* __restrict__ edst,
                                                   float* __restrict__ e_out,
                                                   const unsigned* __restrict__ m_bits,
                                                   float* __restrict__ denom) {
  int i = blockIdx.x * 256 + threadIdx.x;
  if (i >= ETOT) return;
  int t = (i < EE) ? edst[i] : (i - EE);
  float w = expf(e_out[i] - ord2f(m_bits[t]));
  e_out[i] = w;
  atomicAdd(&denom[t], w);
}

// Pass 3: one wave per edge scatters alpha * xl[src] into out[dst]
__global__ __launch_bounds__(256) void edge_scatter(const int* __restrict__ esrc,
                                                    const int* __restrict__ edst,
                                                    const float* __restrict__ e_out,
                                                    const float* __restrict__ denom,
                                                    const float* __restrict__ xl,
                                                    float* __restrict__ out) {
  int wave = threadIdx.x >> 5, lane = threadIdx.x & 31;
  int i = blockIdx.x * 8 + wave;
  if (i >= ETOT) return;
  int s, t;
  if (i < EE) { s = esrc[i]; t = edst[i]; } else { s = t = i - EE; }
  float alpha = e_out[i] / (denom[t] + 1e-16f);
  const float* xs = xl + (size_t)s * 256;
  float* o = out + (size_t)t * 256;
  for (int j = 0; j < 8; ++j) {
    int d = lane + 32 * j;
    atomicAdd(&o[d], alpha * xs[d]);
  }
}

// ---------------------------------------------------------------------------
// Host orchestration
// ---------------------------------------------------------------------------
extern "C" void kernel_launch(void* const* d_in, const int* in_sizes, int n_in,
                              void* d_out, int out_size, void* d_ws, size_t ws_size,
                              hipStream_t stream) {
  (void)in_sizes; (void)n_in; (void)out_size; (void)ws_size;

  const float* x     = (const float*)d_in[0];
  const int*   edges = (const int*)d_in[1];
  const float* W1 = (const float*)d_in[2];  const float* b1  = (const float*)d_in[3];
  const float* W2 = (const float*)d_in[4];  const float* b2  = (const float*)d_in[5];
  const float* W3 = (const float*)d_in[6];  const float* b3  = (const float*)d_in[7];
  const float* g1 = (const float*)d_in[8];  const float* be1 = (const float*)d_in[9];
  const float* g2 = (const float*)d_in[10]; const float* be2 = (const float*)d_in[11];
  const float* g3 = (const float*)d_in[12]; const float* be3 = (const float*)d_in[13];
  const float* Wg1 = (const float*)d_in[14];
  const float* as1 = (const float*)d_in[15]; const float* ad1 = (const float*)d_in[16];
  const float* bg1 = (const float*)d_in[17];
  const float* g4  = (const float*)d_in[18]; const float* be4 = (const float*)d_in[19];
  const float* Wg2 = (const float*)d_in[20];
  const float* as2 = (const float*)d_in[21]; const float* ad2 = (const float*)d_in[22];
  const float* bg2 = (const float*)d_in[23];
  const float* g5  = (const float*)d_in[24]; const float* be5 = (const float*)d_in[25];

  const int* esrc = edges;
  const int* edst = edges + EE;

  // --- carve workspace (~135 MB) ---
  char* ws = (char*)d_ws;
  size_t off = 0;
  auto take = [&](size_t bytes) -> char* {
    char* p = ws + off;
    off += (bytes + 255) & ~(size_t)255;
    return p;
  };
  unsigned short* W1b  = (unsigned short*)take((size_t)FIN * DD * 2);
  unsigned short* W2b  = (unsigned short*)take((size_t)DD * DD * 2);
  unsigned short* W3b  = (unsigned short*)take((size_t)DD * DD * 2);
  unsigned short* Wg1b = (unsigned short*)take((size_t)DD * DD * 2);
  unsigned short* Wg2b = (unsigned short*)take((size_t)DD * DD * 2);
  float*          pre   = (float*)take((size_t)NN * DD * 4);
  unsigned short* act16 = (unsigned short*)take((size_t)NN * DD * 2);
  float*          xl    = (float*)take((size_t)NN * DD * 4);
  float*    a_src  = (float*)take((size_t)NN * 4);
  float*    a_dst  = (float*)take((size_t)NN * 4);
  unsigned* m_bits = (unsigned*)take((size_t)NN * 4);
  float*    denom  = (float*)take((size_t)NN * 4);
  float*    w_edge = (float*)take((size_t)ETOT * 4);
  float*    psum   = (float*)take((size_t)CH * DD * 4);
  float*    psq    = (float*)take((size_t)CH * DD * 4);
  float*    scale  = (float*)take((size_t)DD * 4);
  float*    shift  = (float*)take((size_t)DD * 4);

  const int ND      = NN * DD;
  const int NDBLK   = (ND + 255) / 256;
  const int MB      = (NN + 63) / 64;        // 782 GEMM blocks
  const int EBLK    = (ETOT + 255) / 256;
  const int ESCBLK  = (ETOT + 7) / 8;
  const int NODEBLK = (NN + 7) / 8;

  // --- weights to bf16 ---
  {
    int n1 = FIN * DD, nd = DD * DD;
    f32_to_bf16_k<<<(n1 + 255) / 256, 256, 0, stream>>>(W1, W1b, n1);
    f32_to_bf16_k<<<(nd + 255) / 256, 256, 0, stream>>>(W2, W2b, nd);
    f32_to_bf16_k<<<(nd + 255) / 256, 256, 0, stream>>>(W3, W3b, nd);
    f32_to_bf16_k<<<(nd + 255) / 256, 256, 0, stream>>>(Wg1, Wg1b, nd);
    f32_to_bf16_k<<<(nd + 255) / 256, 256, 0, stream>>>(Wg2, Wg2b, nd);
  }

  auto bn_stage = [&](const float* gamma, const float* beta,
                      float* outf, unsigned short* outb) {
    col_stats_partial<<<CH, 256, 0, stream>>>(pre, psum, psq);
    col_stats_final<<<1, 256, 0, stream>>>(psum, psq, gamma, beta, scale, shift);
    bn_elu<<<NDBLK, 256, 0, stream>>>(pre, scale, shift, outf, outb, ND);
  };

  auto gat_stage = [&](const unsigned short* Wgb, const float* att_s,
                       const float* att_d, const float* bias) {
    gemm_kernel<true><<<MB, 256, 0, stream>>>(act16, Wgb, nullptr, xl, NN, DD);
    dots_kernel<<<NODEBLK, 256, 0, stream>>>(xl, att_s, att_d, a_src, a_dst,
                                             m_bits, denom);
    fill_bias<<<NDBLK, 256, 0, stream>>>(pre, bias, ND);
    edge_max<<<EBLK, 256, 0, stream>>>(esrc, edst, a_src, a_dst, w_edge, m_bits);
    edge_expsum<<<EBLK, 256, 0, stream>>>(edst, w_edge, m_bits, denom);
    edge_scatter<<<ESCBLK, 256, 0, stream>>>(esrc, edst, w_edge, denom, xl, pre);
  };

  // --- Layer 1: x @ W1 + b1 -> BN -> ELU (bf16 forward copy only) ---
  gemm_kernel<false><<<MB, 256, 0, stream>>>(x, W1b, b1, pre, NN, FIN);
  bn_stage(g1, be1, nullptr, act16);

  // --- Layers 2, 3 ---
  gemm_kernel<true><<<MB, 256, 0, stream>>>(act16, W2b, b2, pre, NN, DD);
  bn_stage(g2, be2, nullptr, act16);
  gemm_kernel<true><<<MB, 256, 0, stream>>>(act16, W3b, b3, pre, NN, DD);
  bn_stage(g3, be3, nullptr, act16);

  // --- GATConv 1 -> BN -> ELU ---
  gat_stage(Wg1b, as1, ad1, bg1);
  bn_stage(g4, be4, nullptr, act16);

  // --- GATConv 2 -> BN -> ELU -> d_out (fp32) ---
  gat_stage(Wg2b, as2, ad2, bg2);
  bn_stage(g5, be5, (float*)d_out, nullptr);
}
